// Message_12395275616824
// MI455X (gfx1250) — matrix operand order repaired and checked by reading, not compiled, not run on hardware
//
#include <hip/hip_runtime.h>

// MI455X / gfx1250: wave32, WMMA f32<-f16 16x16x32 path.
// Memory-bound MLP: ~460 MB traffic @ 23.3 TB/s ~= 20 us floor; 28 GFLOP of
// matrix work is ~5 us on the f16 WMMA path, so we stage f32->f16 and use
// v_wmma_f32_16x16x32_f16 with f32 accumulation. Layer-1 bias is folded into
// the K-padding (feature 68 == 1.0, W1 row 68 == b1), and each wave prefetches
// the next tile's h block (global_prefetch) to hide HBM latency behind the
// WMMA + LDS-relayout chain.

typedef __attribute__((ext_vector_type(16))) _Float16 v16h;
typedef __attribute__((ext_vector_type(8)))  float    v8f;

#define WAVES_PER_BLOCK 8
#define THREADS (WAVES_PER_BLOCK * 32)

// 16-bit A-matrix 16x32 k order used here (contiguous, not ISA table order): lane m=l&15, hi=l>>4,
// VGPR p (pair), half t: k = hi*16 + 2p + t
__device__ __forceinline__ int a_k0(int p, int hi) {
  return hi * 16 + 2 * p;
}

// Load one A fragment (16x32 f16) for this lane from an LDS row (k pairs are
// contiguous -> aligned b32 LDS loads; compiler merges them into ds_load_b128).
__device__ __forceinline__ v16h load_a_frag(const _Float16* xrow, int kbase, int hi) {
  union { v16h v; unsigned u[8]; } r;
#pragma unroll
  for (int p = 0; p < 8; ++p) {
    int k = kbase + a_k0(p, hi);
    r.u[p] = *(const unsigned*)(xrow + k);
  }
  return r.v;
}

// 16-bit B-matrix 32x16 k order used here (contiguous, same as A): lane n=l&15, hi=l>>4,
// VGPR p half t: k = hi*16 + 2p + t. Built once from f32 weights (row-major
// [K][ldn]); k == Ktot optionally sources a bias row (for the augmented-1
// feature trick); other out-of-range k or invalid n are zero.
__device__ __forceinline__ v16h load_b_frag(const float* __restrict__ W, int ldn,
                                            int kbase, int n, int Ktot, bool nvalid,
                                            const float* __restrict__ biasRow,
                                            int hi) {
  union { v16h v; _Float16 h[16]; } r;
#pragma unroll
  for (int p = 0; p < 8; ++p) {
#pragma unroll
    for (int t = 0; t < 2; ++t) {
      int k = kbase + hi * 16 + 2 * p + t;
      float w = 0.0f;
      if (nvalid) {
        if (k < Ktot) w = W[k * ldn + n];
        else if (k == Ktot && biasRow) w = biasRow[n];
      }
      r.h[2 * p + t] = (_Float16)w;
    }
  }
  return r.v;
}

__global__ __launch_bounds__(THREADS) void mlp_msg_kernel(
    const float* __restrict__ h, const float* __restrict__ J,
    const float* __restrict__ b_in, const float* __restrict__ b_out,
    const float* __restrict__ W1, const float* __restrict__ b1,
    const float* __restrict__ W2, const float* __restrict__ b2,
    const float* __restrict__ W3, const float* __restrict__ b3,
    float* __restrict__ out, int nRows, int numChunks) {
  // Per-wave private LDS slices (no cross-wave sharing).
  __shared__ __align__(16) _Float16 ldsX[WAVES_PER_BLOCK * 16 * 96];  // x, K padded to 96
  __shared__ __align__(16) _Float16 ldsY[WAVES_PER_BLOCK * 16 * 64];  // hidden activations

  const int lane = threadIdx.x & 31;
  const int wave = threadIdx.x >> 5;
  const int nn = lane & 15;
  const int hi = lane >> 4;
  const int m = nn;  // A-fragment row for this lane

  _Float16* xs = &ldsX[wave * 16 * 96];
  _Float16* ys = &ldsY[wave * 16 * 64];

  // ---- Hoisted weight fragments (reused for every row tile) ----
  // Layer 1: K augmented to 69 (row 68 = b1, paired with x feature 68 == 1.0).
  v16h B1f[3][4], B2f[2][4], B3f[2];
  float b2v[4];
#pragma unroll
  for (int nt = 0; nt < 4; ++nt) {
    int n = nt * 16 + nn;
#pragma unroll
    for (int kt = 0; kt < 3; ++kt)
      B1f[kt][nt] = load_b_frag(W1, 64, kt * 32, n, 68, true, b1, hi);
#pragma unroll
    for (int kt = 0; kt < 2; ++kt)
      B2f[kt][nt] = load_b_frag(W2, 64, kt * 32, n, 64, true, nullptr, hi);
    b2v[nt] = b2[n];
  }
  const bool n5 = (nn < 5);
#pragma unroll
  for (int kt = 0; kt < 2; ++kt)
    B3f[kt] = load_b_frag(W3, 5, kt * 32, nn, 64, n5, nullptr, hi);
  const float b3v = n5 ? b3[nn] : 0.0f;

  // Uniform trip count per block (chunks depend only on blockIdx).
  for (int chunk = blockIdx.x; chunk < numChunks; chunk += gridDim.x) {
    const int tile = chunk * WAVES_PER_BLOCK + wave;
    const long row0 = (long)tile * 16;

    // Prefetch next chunk's h tile: 16 rows x 256B = 4KB = 32 lanes x 128B.
    {
      int nextTile = (chunk + (int)gridDim.x) * WAVES_PER_BLOCK + wave;
      long nrow0 = (long)nextTile * 16;
      if (nrow0 + 16 <= nRows)
        __builtin_prefetch(h + nrow0 * 64 + lane * 32, 0, 3);
    }

    // ---- stage x = [h | b_in b_out J -J | 1.0 | zero pad] as f16, 16x96 ----
#pragma unroll
    for (int i = 0; i < 8; ++i) {
      int idx = i * 32 + lane;           // 256 float4 chunks of h per tile
      int r = idx >> 4;
      int c = (idx & 15) << 2;
      long row = row0 + r;
      float4 g = make_float4(0.f, 0.f, 0.f, 0.f);
      if (row < nRows) g = *(const float4*)(h + row * 64 + c);
      union { _Float16 q[4]; uint2 u; } pk;
      pk.q[0] = (_Float16)g.x; pk.q[1] = (_Float16)g.y;
      pk.q[2] = (_Float16)g.z; pk.q[3] = (_Float16)g.w;
      *(uint2*)(&xs[r * 96 + c]) = pk.u;
    }
    if (lane < 16) {
      long row = row0 + lane;
      float bi = 0.f, bo = 0.f, Jv = 0.f;
      if (row < nRows) { bi = b_in[row]; bo = b_out[row]; Jv = J[row]; }
      union { _Float16 q[4]; uint2 u; } pk;
      pk.q[0] = (_Float16)bi; pk.q[1] = (_Float16)bo;
      pk.q[2] = (_Float16)Jv; pk.q[3] = (_Float16)(-Jv);
      *(uint2*)(&xs[lane * 96 + 64]) = pk.u;
      union { _Float16 q[4]; uint2 u; } one;
      one.q[0] = (_Float16)1.0f; one.q[1] = (_Float16)0.0f;
      one.q[2] = (_Float16)0.0f; one.q[3] = (_Float16)0.0f;
      *(uint2*)(&xs[lane * 96 + 68]) = one.u;   // augmented-1 feature for b1
      uint2 z; z.x = 0u; z.y = 0u;
#pragma unroll
      for (int c = 72; c < 96; c += 4) *(uint2*)(&xs[lane * 96 + c]) = z;
    }
    __syncthreads();

    // ---- layer 1: [16x96(f16)] x [96x64 (incl. bias row)] -> ReLU -> ys ----
    v16h a0 = load_a_frag(xs + m * 96, 0, hi);
    v16h a1 = load_a_frag(xs + m * 96, 32, hi);
    v16h a2 = load_a_frag(xs + m * 96, 64, hi);
#pragma unroll
    for (int nt = 0; nt < 4; ++nt) {
      v8f c = {};
      c = __builtin_amdgcn_wmma_f32_16x16x32_f16(false, a0, false, B1f[0][nt], (short)0, c, false, false);
      c = __builtin_amdgcn_wmma_f32_16x16x32_f16(false, a1, false, B1f[1][nt], (short)0, c, false, false);
      c = __builtin_amdgcn_wmma_f32_16x16x32_f16(false, a2, false, B1f[2][nt], (short)0, c, false, false);
#pragma unroll
      for (int r = 0; r < 8; ++r) {
        float v = c[r];
        v = v > 0.f ? v : 0.f;
        ys[(r + 8 * hi) * 64 + nt * 16 + nn] = (_Float16)v;
      }
    }

    // ---- layer 2: [16x64] x [64x64] -> ReLU -> ys (loads precede stores) ----
    v16h p0 = load_a_frag(ys + m * 64, 0, hi);
    v16h p1 = load_a_frag(ys + m * 64, 32, hi);
#pragma unroll
    for (int nt = 0; nt < 4; ++nt) {
      v8f c = {};
      c = __builtin_amdgcn_wmma_f32_16x16x32_f16(false, p0, false, B2f[0][nt], (short)0, c, false, false);
      c = __builtin_amdgcn_wmma_f32_16x16x32_f16(false, p1, false, B2f[1][nt], (short)0, c, false, false);
#pragma unroll
      for (int r = 0; r < 8; ++r) {
        float v = c[r] + b2v[nt];
        v = v > 0.f ? v : 0.f;
        ys[(r + 8 * hi) * 64 + nt * 16 + nn] = (_Float16)v;
      }
    }

    // ---- layer 3: [16x64] x [64x16(5 valid)] + b3 -> out ----
    v16h q0 = load_a_frag(ys + m * 64, 0, hi);
    v16h q1 = load_a_frag(ys + m * 64, 32, hi);
    {
      v8f c = {};
      c = __builtin_amdgcn_wmma_f32_16x16x32_f16(false, q0, false, B3f[0], (short)0, c, false, false);
      c = __builtin_amdgcn_wmma_f32_16x16x32_f16(false, q1, false, B3f[1], (short)0, c, false, false);
      if (n5) {
#pragma unroll
        for (int r = 0; r < 8; ++r) {
          long row = row0 + r + 8 * hi;
          if (row < nRows) out[row * 5 + nn] = c[r] + b3v;
        }
      }
    }
    __syncthreads();
  }
}

extern "C" void kernel_launch(void* const* d_in, const int* in_sizes, int n_in,
                              void* d_out, int out_size, void* d_ws, size_t ws_size,
                              hipStream_t stream) {
  const float* h     = (const float*)d_in[0];
  const float* J     = (const float*)d_in[1];
  const float* b_in  = (const float*)d_in[2];
  const float* b_out = (const float*)d_in[3];
  const float* W1    = (const float*)d_in[4];
  const float* b1    = (const float*)d_in[5];
  const float* W2    = (const float*)d_in[6];
  const float* b2    = (const float*)d_in[7];
  const float* W3    = (const float*)d_in[8];
  const float* b3    = (const float*)d_in[9];
  float* out = (float*)d_out;

  int nRows = in_sizes[1];                     // J is [B,E,1] -> B*E rows
  int T = (nRows + 15) / 16;                   // 16-row tiles
  int numChunks = (T + WAVES_PER_BLOCK - 1) / WAVES_PER_BLOCK;
  int blocks = numChunks < 1024 ? numChunks : 1024;

  mlp_msg_kernel<<<blocks, THREADS, 0, stream>>>(h, J, b_in, b_out, W1, b1, W2, b2,
                                                 W3, b3, out, nRows, numChunks);
}
